// VoxelResBackBone8x_72971494359432
// MI455X (gfx1250) — matrix operand order repaired and checked by reading, not compiled, not run on hardware
//
#include <hip/hip_runtime.h>
#include <hip/hip_bf16.h>

// ---------------------------------------------------------------------------
// VoxelResBackBone8x on gfx1250: fp32 grids, bf16 WMMA (v_wmma_f32_16x16x32_bf16)
// implicit-GEMM convolutions. Per-block B-fragment staging through LDS via the
// Tensor Data Mover (tensor_load_to_lds + s_wait_tensorcnt), masked-BN via LDS
// reductions + f32 atomics. Conv kernel templated on <K-chunks, N-tiles> so the
// inner GEMM unrolls into KC*NT back-to-back WMMAs sharing one A fragment.
// ---------------------------------------------------------------------------

typedef __attribute__((ext_vector_type(16))) __bf16 v16bf;
typedef __attribute__((ext_vector_type(8)))  float  v8f;
typedef unsigned int u32x4 __attribute__((ext_vector_type(4)));
typedef int i32x8 __attribute__((ext_vector_type(8)));
typedef int i32x4 __attribute__((ext_vector_type(4)));

struct ConvGeom {
  int Din, Hin, Win, Cin;
  int Dout, Hout, Wout, Cout;
  int kd, kh, kw;
  int sd, sh, sw;
  int pd, ph, pw;
  int dd, dh, dw;
  int hwShift, wShift;  // log2(Hout*Wout), log2(Wout) -- all powers of two
};

__global__ void k_scatter(const float* __restrict__ feat,
                          const int* __restrict__ coords,
                          float* __restrict__ grid, float* __restrict__ cnt,
                          int n, int H, int W) {
  int i = blockIdx.x * blockDim.x + threadIdx.x;
  if (i >= n) return;
  int z = coords[i * 4 + 1], y = coords[i * 4 + 2], x = coords[i * 4 + 3];
  int v = (z * H + y) * W + x;
  #pragma unroll
  for (int c = 0; c < 4; ++c) atomicAdd(&grid[v * 4 + c], feat[i * 4 + c]);
  atomicAdd(&cnt[v], 1.0f);
}

__global__ void k_binarize(float* __restrict__ m, int n) {
  int i = blockIdx.x * blockDim.x + threadIdx.x;
  if (i < n) m[i] = (m[i] > 0.0f) ? 1.0f : 0.0f;
}

// repack fp32 [tap][cin][cout] weights into bf16 WMMA B-fragments:
// [tap][kchunk][nchunk][lane 0..31][16 bf16]; B(32x16): lane l: n=l&15,
// element e -> k = (l>>4)*16 + e
__global__ void k_repack(const float* __restrict__ w, unsigned short* __restrict__ dst,
                         int taps, int cin, int cout) {
  int kc = (cin + 31) >> 5, nt = cout >> 4;
  int frags = taps * kc * nt;
  int t = blockIdx.x * blockDim.x + threadIdx.x;
  if (t >= frags * 32) return;
  int lane = t & 31;
  int f = t >> 5;
  int tn = f % nt;
  int f2 = f / nt;
  int c = f2 % kc;
  int tap = f2 / kc;
  int hf = lane >> 4, l15 = lane & 15;
  int n = tn * 16 + l15;
  unsigned short* o = dst + f * 512 + lane * 16;
  #pragma unroll
  for (int e = 0; e < 16; ++e) {
    int k = c * 32 + hf * 16 + e;
    float val = (k < cin) ? w[(tap * cin + k) * cout + n] : 0.0f;
    __bf16 h = (__bf16)val;
    o[e] = __builtin_bit_cast(unsigned short, h);
  }
}

// TDM: copy one tap's B-fragment block (KC rows x NT KB, row stride nTilesTot KB)
// from global into LDS. Issued by wave 0 only; completion via TENSORcnt.
__device__ __forceinline__ void tdm_load_b(const unsigned short* gsrc,
                                           unsigned ldsByteAddr, int KC, int NT,
                                           int nTilesTot) {
  unsigned long long ga = (unsigned long long)gsrc;
  u32x4 g0 = {1u,                               // count=1, user descriptor
              ldsByteAddr,                      // lds_addr
              (unsigned)ga,                     // global_addr[31:0]
              (unsigned)((ga >> 32) & 0x1FFFFFFu) | (2u << 30)};  // type=2
  unsigned td0 = (unsigned)nTilesTot * 256u;    // tensor_dim0 (4B units)
  unsigned td1 = 1u << 20;                      // tensor_dim1 (generous)
  unsigned tile0 = (unsigned)NT * 256u;         // tile_dim0
  unsigned tile1 = (unsigned)KC;                // tile_dim1
  unsigned long long st0 = (unsigned long long)nTilesTot * 256ull;
  i32x8 g1 = {(int)(2u << 16),                                   // data_size=4B
              (int)((td0 & 0xffffu) << 16),                      // td0 lo
              (int)((td0 >> 16) | ((td1 & 0xffffu) << 16)),      // td0 hi, td1 lo
              (int)((td1 >> 16) | (tile0 << 16)),                // td1 hi, tile0
              (int)tile1,                                        // tile1, tile2=0
              (int)(st0 & 0xffffffffull),                        // stride0 lo
              (int)(st0 >> 32),                                  // stride0 hi
              0};
  i32x4 gz = {0, 0, 0, 0};
  i32x8 gz8 = {0, 0, 0, 0, 0, 0, 0, 0};
  __builtin_amdgcn_tensor_load_to_lds(g0, g1, gz, gz, gz8, 0);
}

// implicit-GEMM conv: one wave = 16 output voxels x (NT*16) output channels.
// blockIdx.y selects the N-group so all 8 waves of a block share B fragments.
// A(16x32 bf16): lane l: m=l&15; element e ->
//   k = (l>>4)*16 + e  => one contiguous 16-channel run
template <int KC, int NT>
__global__ void __launch_bounds__(256)
k_conv_wmma(const float* __restrict__ in, const unsigned short* __restrict__ wfrag,
            float* __restrict__ out, ConvGeom g) {
  __shared__ unsigned short ldsb[KC * NT * 512];
  const int lane = threadIdx.x & 31;
  const int wave = threadIdx.x >> 5;
  const int nvox = g.Dout * g.Hout * g.Wout;
  const int mTiles = (nvox + 15) >> 4;
  const int nTilesTot = g.Cout >> 4;
  const int tm = blockIdx.x * (blockDim.x >> 5) + wave;
  const int tn = blockIdx.y * NT;
  const bool active = tm < mTiles;
  const int hf = lane >> 4;
  const int l15 = lane & 15;

  const int vox = tm * 16 + l15;
  const bool vvalid = active && vox < nvox;
  const int zo = vox >> g.hwShift;
  const int rr = vox & ((1 << g.hwShift) - 1);
  const int yo = rr >> g.wShift;
  const int xo = rr & ((1 << g.wShift) - 1);

  v8f acc[NT];
  #pragma unroll
  for (int t = 0; t < NT; ++t) acc[t] = (v8f){};

  const unsigned ldsAddr = (unsigned)(unsigned long long)(&ldsb[0]);

  int tap = 0;
  for (int kz = 0; kz < g.kd; ++kz)
    for (int ky = 0; ky < g.kh; ++ky)
      for (int kx = 0; kx < g.kw; ++kx, ++tap) {
        // stage this tap's weight fragments into LDS via the Tensor Data Mover
        __syncthreads();
        if (wave == 0) {
          const unsigned short* gsrc =
              wfrag + (tap * KC * nTilesTot + tn) * 512;
          tdm_load_b(gsrc, ldsAddr, KC, NT, nTilesTot);
          __builtin_amdgcn_s_wait_tensorcnt(0);
        }
        __syncthreads();

        const int zi = zo * g.sd - g.pd + kz * g.dd;
        const int yi = yo * g.sh - g.ph + ky * g.dh;
        const int xi = xo * g.sw - g.pw + kx * g.dw;
        const bool inb = vvalid && zi >= 0 && zi < g.Din && yi >= 0 &&
                         yi < g.Hin && xi >= 0 && xi < g.Win;
        const float* base = in + ((zi * g.Hin + yi) * g.Win + xi) * g.Cin;
        if (inb) __builtin_prefetch(base, 0, 1);  // global_prefetch_b8
        #pragma unroll
        for (int c = 0; c < KC; ++c) {
          v16bf a;
          #pragma unroll
          for (int r = 0; r < 2; ++r) {
            const int c0 = c * 32 + hf * 16 + r * 8;
            if (inb && (c0 + 8 <= g.Cin)) {
              const float4 f0 = *(const float4*)(base + c0);
              const float4 f1 = *(const float4*)(base + c0 + 4);
              a[r * 8 + 0] = (__bf16)f0.x; a[r * 8 + 1] = (__bf16)f0.y;
              a[r * 8 + 2] = (__bf16)f0.z; a[r * 8 + 3] = (__bf16)f0.w;
              a[r * 8 + 4] = (__bf16)f1.x; a[r * 8 + 5] = (__bf16)f1.y;
              a[r * 8 + 6] = (__bf16)f1.z; a[r * 8 + 7] = (__bf16)f1.w;
            } else {
              #pragma unroll
              for (int j = 0; j < 8; ++j) {
                float f = 0.0f;
                if (inb && (c0 + j) < g.Cin) f = base[c0 + j];
                a[r * 8 + j] = (__bf16)f;
              }
            }
          }
          #pragma unroll
          for (int t = 0; t < NT; ++t) {
            const v16bf b = *(const v16bf*)&ldsb[(c * NT + t) * 512 + lane * 16];
            acc[t] = __builtin_amdgcn_wmma_f32_16x16x32_bf16(
                false, a, false, b, (short)0, acc[t], false, false);
          }
        }
      }

  // D(16x16 f32): lane l, vgpr r -> m = (l>>4)*8 + r, n = l&15
  if (active) {
    #pragma unroll
    for (int r = 0; r < 8; ++r) {
      const int v2 = tm * 16 + hf * 8 + r;
      if (v2 < nvox) {
        #pragma unroll
        for (int t = 0; t < NT; ++t)
          out[v2 * g.Cout + (tn + t) * 16 + l15] = acc[t][r];
      }
    }
  }
}

__global__ void k_dilate(const float* __restrict__ mi, float* __restrict__ mo,
                         ConvGeom g) {
  int nvox = g.Dout * g.Hout * g.Wout;
  int i = blockIdx.x * blockDim.x + threadIdx.x;
  if (i >= nvox) return;
  int zo = i >> g.hwShift;
  int r = i & ((1 << g.hwShift) - 1);
  int yo = r >> g.wShift, xo = r & ((1 << g.wShift) - 1);
  float v = 0.0f;
  for (int kz = 0; kz < g.kd; ++kz)
    for (int ky = 0; ky < g.kh; ++ky)
      for (int kx = 0; kx < g.kw; ++kx) {
        int zi = zo * g.sd - g.pd + kz * g.dd;
        int yi = yo * g.sh - g.ph + ky * g.dh;
        int xi = xo * g.sw - g.pw + kx * g.dw;
        if (zi >= 0 && zi < g.Din && yi >= 0 && yi < g.Hin && xi >= 0 &&
            xi < g.Win)
          v = fmaxf(v, mi[(zi * g.Hin + yi) * g.Win + xi]);
      }
  mo[i] = (v > 0.0f) ? 1.0f : 0.0f;
}

__global__ void k_count(const float* __restrict__ m, int n, float* __restrict__ c) {
  float s = 0.0f;
  for (int i = blockIdx.x * blockDim.x + threadIdx.x; i < n;
       i += gridDim.x * blockDim.x)
    s += m[i];
  __shared__ float sh[256];
  sh[threadIdx.x] = s;
  __syncthreads();
  for (int o = 128; o > 0; o >>= 1) {
    if (threadIdx.x < (unsigned)o) sh[threadIdx.x] += sh[threadIdx.x + o];
    __syncthreads();
  }
  if (threadIdx.x == 0) atomicAdd(c, sh[0]);
}

__global__ void k_bn_stats(const float* __restrict__ x, const float* __restrict__ m,
                           int nvox, int cout, float* __restrict__ stats) {
  int c = blockIdx.x;
  float s = 0.0f, ss = 0.0f;
  for (int v = blockIdx.y * blockDim.x + threadIdx.x; v < nvox;
       v += gridDim.y * blockDim.x) {
    if (m[v] != 0.0f) {
      float f = x[v * cout + c];
      s += f;
      ss += f * f;
    }
  }
  __shared__ float sh[512];
  sh[threadIdx.x] = s;
  sh[256 + threadIdx.x] = ss;
  __syncthreads();
  for (int o = 128; o > 0; o >>= 1) {
    if (threadIdx.x < (unsigned)o) {
      sh[threadIdx.x] += sh[threadIdx.x + o];
      sh[256 + threadIdx.x] += sh[256 + threadIdx.x + o];
    }
    __syncthreads();
  }
  if (threadIdx.x == 0) {
    atomicAdd(&stats[c], sh[0]);
    atomicAdd(&stats[cout + c], sh[256]);
  }
}

__global__ void k_bn_finalize(const float* __restrict__ stats,
                              const float* __restrict__ cnt,
                              const float* __restrict__ gamma,
                              const float* __restrict__ beta,
                              float* __restrict__ ss, int cout) {
  int c = blockIdx.x * blockDim.x + threadIdx.x;
  if (c >= cout) return;
  float n = fmaxf(cnt[0], 1.0f);
  float mean = stats[c] / n;
  float var = stats[cout + c] / n - mean * mean;
  float inv = rsqrtf(var + 1e-3f);
  float sc = gamma[c] * inv;
  ss[c] = sc;
  ss[cout + c] = beta[c] - mean * sc;
}

// y = relu(scale*x + shift [+ res]) * mask   (conv bias cancels under BN)
__global__ void k_bn_apply(const float* __restrict__ x, const float* __restrict__ res,
                           const float* __restrict__ m, const float* __restrict__ ss,
                           float* __restrict__ out, int nvox, int cout) {
  int total = nvox * cout;
  for (int i = blockIdx.x * blockDim.x + threadIdx.x; i < total;
       i += gridDim.x * blockDim.x) {
    int v = i / cout;
    int c = i % cout;
    float f = x[i] * ss[c] + ss[cout + c];
    if (res) f += res[i];
    out[i] = fmaxf(f, 0.0f) * m[v];
  }
}

__global__ void k_add3(const float* a, const float* b, const float* c, float* o,
                       int n) {
  int i = blockIdx.x * blockDim.x + threadIdx.x;
  if (i < n) o[i] = a[i] + b[i] + c[i];
}

__global__ void k_max3(const float* a, const float* b, const float* c, float* o,
                       int n) {
  int i = blockIdx.x * blockDim.x + threadIdx.x;
  if (i < n) o[i] = (fmaxf(fmaxf(a[i], b[i]), c[i]) > 0.0f) ? 1.0f : 0.0f;
}

__global__ void k_hcompress(const float* __restrict__ x, const float* __restrict__ m,
                            float* __restrict__ x2d, float* __restrict__ m2d,
                            int Dz, int HW, int C) {
  int total = HW * C;
  int i = blockIdx.x * blockDim.x + threadIdx.x;
  if (i >= total) return;
  int hw = i / C, c = i % C;
  float s = 0.0f;
  for (int z = 0; z < Dz; ++z) s += x[(z * HW + hw) * C + c];
  x2d[i] = s;
  if (c == 0) {
    float mm = 0.0f;
    for (int z = 0; z < Dz; ++z) mm = fmaxf(mm, m[z * HW + hw]);
    m2d[hw] = (mm > 0.0f) ? 1.0f : 0.0f;
  }
}

// ===========================================================================
extern "C" void kernel_launch(void* const* d_in, const int* in_sizes, int n_in,
                              void* d_out, int out_size, void* d_ws, size_t ws_size,
                              hipStream_t stream) {
  (void)n_in; (void)out_size; (void)ws_size;
  const float* feat = (const float*)d_in[0];
  const int* coords = (const int*)d_in[1];
  const int N = in_sizes[0] / 4;

  // Param leaves: JAX sorted-key tree-flatten order after the 2 array inputs.
  // blk leaves: +0 b1, +1 b2, +2 bt1, +3 bt2, +4 g1, +5 g2, +6 w1, +7 w2
  // stage: blocks at S, S+8; down at S+16 (b), S+17 (g), S+18 (w)
  const int C1 = 2, C2 = 18, C3 = 37, C4 = 56, C4A = 75, C4B = 94;
  const int CI_B = 113, CI_G = 114, CI_W = 115;
  const int SH_B1 = 116, SH_B2 = 117, SH_G1 = 118, SH_G2 = 119, SH_W1 = 120,
            SH_W2 = 121;
  auto P = [&](int i) -> const float* { return (const float*)d_in[i]; };

  const int D0 = 21, H0 = 256, W0 = 256;
  const long NV0 = (long)D0 * H0 * W0;
  const long NV2 = 11L * 128 * 128;
  const long NV3 = 6L * 64 * 64;
  const long NV4 = 2L * 32 * 32;
  const int HW4 = 1024;

  float* wsf = (float*)d_ws;
  size_t pos = 0;
  auto allocF = [&](size_t n) -> float* {
    float* p = wsf + pos;
    pos += (n + 63) & ~(size_t)63;
    return p;
  };
  auto allocW = [&](size_t nu16) -> unsigned short* {
    return (unsigned short*)allocF((nu16 + 1) / 2);
  };

  float* stats = allocF(256);
  float* ssb = allocF(256);
  float* cntb = allocF(64);
  float* grid0 = allocF(NV0 * 4);
  float* mask0 = allocF(NV0);
  float* m2 = allocF(NV2);
  float* m3 = allocF(NV3);
  float* m4 = allocF(NV4);
  float* m4a = allocF(NV4);
  float* m4b = allocF(NV4);
  float* msu = allocF(NV4);
  float* m2d = allocF(HW4);
  float* m2d1 = allocF(HW4);
  float* m2d2 = allocF(HW4);
  float* bufA = allocF(NV0 * 16);
  float* bufB = allocF(NV0 * 16);
  float* bufC = allocF(NV0 * 16);
  float* s1 = allocF(NV4 * 64);
  float* s2 = allocF(NV4 * 64);
  float* s3 = allocF(NV4 * 64);
  float* b4o0 = allocF(NV4 * 64);
  float* b4o1 = allocF(NV4 * 64);
  float* b4o2 = allocF(NV4 * 64);
  float* xs = allocF(NV4 * 64);
  float* x2d = allocF((size_t)HW4 * 64);
  float* y2d = allocF((size_t)HW4 * 128);
  float* t2d = allocF((size_t)HW4 * 128);

  auto repack = [&](const float* w, int taps, int cin, int cout) -> unsigned short* {
    int kc = (cin + 31) / 32, nt = cout / 16;
    size_t frags = (size_t)taps * kc * nt;
    unsigned short* dst = allocW(frags * 512);
    long threads = (long)frags * 32;
    k_repack<<<(int)((threads + 255) / 256), 256, 0, stream>>>(w, dst, taps, cin, cout);
    return dst;
  };

  unsigned short* wf_ci = repack(P(CI_W), 27, 4, 16);
  unsigned short* wf_c1b0w1 = repack(P(C1 + 6), 27, 16, 16);
  unsigned short* wf_c1b0w2 = repack(P(C1 + 7), 27, 16, 16);
  unsigned short* wf_c1b1w1 = repack(P(C1 + 8 + 6), 27, 16, 16);
  unsigned short* wf_c1b1w2 = repack(P(C1 + 8 + 7), 27, 16, 16);
  unsigned short* wf_c2d = repack(P(C2 + 18), 27, 16, 32);
  unsigned short* wf_c2b0w1 = repack(P(C2 + 6), 27, 32, 32);
  unsigned short* wf_c2b0w2 = repack(P(C2 + 7), 27, 32, 32);
  unsigned short* wf_c2b1w1 = repack(P(C2 + 8 + 6), 27, 32, 32);
  unsigned short* wf_c2b1w2 = repack(P(C2 + 8 + 7), 27, 32, 32);
  unsigned short* wf_c3d = repack(P(C3 + 18), 27, 32, 64);
  unsigned short* wf_c3b0w1 = repack(P(C3 + 6), 27, 64, 64);
  unsigned short* wf_c3b0w2 = repack(P(C3 + 7), 27, 64, 64);
  unsigned short* wf_c3b1w1 = repack(P(C3 + 8 + 6), 27, 64, 64);
  unsigned short* wf_c3b1w2 = repack(P(C3 + 8 + 7), 27, 64, 64);
  unsigned short* wf_c4d = repack(P(C4 + 18), 27, 64, 64);
  unsigned short* wf_c4b0w1 = repack(P(C4 + 6), 27, 64, 64);
  unsigned short* wf_c4b0w2 = repack(P(C4 + 7), 27, 64, 64);
  unsigned short* wf_c4b1w1 = repack(P(C4 + 8 + 6), 27, 64, 64);
  unsigned short* wf_c4b1w2 = repack(P(C4 + 8 + 7), 27, 64, 64);
  unsigned short* wf_c4ad = repack(P(C4A + 18), 27, 64, 64);
  unsigned short* wf_c4ab0w1 = repack(P(C4A + 6), 27, 64, 64);
  unsigned short* wf_c4ab0w2 = repack(P(C4A + 7), 27, 64, 64);
  unsigned short* wf_c4ab1w1 = repack(P(C4A + 8 + 6), 27, 64, 64);
  unsigned short* wf_c4ab1w2 = repack(P(C4A + 8 + 7), 27, 64, 64);
  unsigned short* wf_c4bd = repack(P(C4B + 18), 75, 64, 64);
  unsigned short* wf_c4bb0w1 = repack(P(C4B + 6), 27, 64, 64);
  unsigned short* wf_c4bb0w2 = repack(P(C4B + 7), 27, 64, 64);
  unsigned short* wf_c4bb1w1 = repack(P(C4B + 8 + 6), 27, 64, 64);
  unsigned short* wf_c4bb1w2 = repack(P(C4B + 8 + 7), 27, 64, 64);
  unsigned short* wf_sh1 = repack(P(SH_W1), 9, 64, 128);
  unsigned short* wf_sh2 = repack(P(SH_W2), 9, 128, 128);

  auto lg2 = [](int v) { int s = 0; while ((1 << s) < v) ++s; return s; };
  auto mkG = [&](int Di, int Hi, int Wi, int Ci, int Do, int Ho, int Wo, int Co,
                 int kd, int kh, int kw, int sd, int sh, int sw, int pd, int ph,
                 int pw, int dd, int dh, int dw) {
    ConvGeom g{Di, Hi, Wi, Ci, Do, Ho, Wo, Co, kd, kh, kw, sd, sh, sw,
               pd, ph, pw, dd, dh, dw, lg2(Ho * Wo), lg2(Wo)};
    return g;
  };
  // dispatch to compile-time specialization: KC = Cin chunks, NT = N tiles/wave
  auto runConv = [&](const float* in, const unsigned short* wf, float* out,
                     ConvGeom g) {
    int nvox = g.Dout * g.Hout * g.Wout;
    int bx = ((nvox + 15) / 16 + 7) / 8;
    int nTiles = g.Cout / 16;
    if (g.Cin <= 32) {
      if (nTiles >= 2) {
        dim3 gr(bx, nTiles / 2);
        k_conv_wmma<1, 2><<<gr, 256, 0, stream>>>(in, wf, out, g);
      } else {
        dim3 gr(bx, nTiles);
        k_conv_wmma<1, 1><<<gr, 256, 0, stream>>>(in, wf, out, g);
      }
    } else if (g.Cin == 64) {
      dim3 gr(bx, nTiles / 2);
      k_conv_wmma<2, 2><<<gr, 256, 0, stream>>>(in, wf, out, g);
    } else {
      dim3 gr(bx, nTiles / 2);
      k_conv_wmma<4, 2><<<gr, 256, 0, stream>>>(in, wf, out, g);
    }
  };
  auto runDilate = [&](const float* mi, float* mo, ConvGeom g) {
    int nvox = g.Dout * g.Hout * g.Wout;
    k_dilate<<<(nvox + 255) / 256, 256, 0, stream>>>(mi, mo, g);
  };
  auto runBN = [&](const float* x, const float* res, const float* mask,
                   const float* gamma, const float* beta, float* out, long nvox,
                   int cout) {
    (void)hipMemsetAsync(stats, 0, 2 * cout * sizeof(float), stream);
    (void)hipMemsetAsync(cntb, 0, sizeof(float), stream);
    k_count<<<64, 256, 0, stream>>>(mask, (int)nvox, cntb);
    dim3 gs(cout, 8);
    k_bn_stats<<<gs, 256, 0, stream>>>(x, mask, (int)nvox, cout, stats);
    k_bn_finalize<<<1, 128, 0, stream>>>(stats, cntb, gamma, beta, ssb, cout);
    long total = nvox * cout;
    int b = (int)((total + 255) / 256);
    if (b > 8192) b = 8192;
    k_bn_apply<<<b, 256, 0, stream>>>(x, res, mask, ssb, out, (int)nvox, cout);
  };
  auto runBlock = [&](const float* x, float* t, float* o,
                      const unsigned short* w1, const unsigned short* w2,
                      const float* mask, int S, long nvox, int C, ConvGeom g) {
    runConv(x, w1, t, g);
    runBN(t, nullptr, mask, P(S + 4), P(S + 2), o, nvox, C);
    runConv(o, w2, t, g);
    runBN(t, x, mask, P(S + 5), P(S + 3), o, nvox, C);
  };

  ConvGeom g_ci  = mkG(21,256,256,4,  21,256,256,16, 3,3,3, 1,1,1, 1,1,1, 1,1,1);
  ConvGeom g_c1  = mkG(21,256,256,16, 21,256,256,16, 3,3,3, 1,1,1, 1,1,1, 1,1,1);
  ConvGeom g_md2 = mkG(21,256,256,1,  11,128,128,1,  3,3,3, 2,2,2, 1,1,1, 1,1,1);
  ConvGeom g_d2  = mkG(21,256,256,16, 11,128,128,32, 3,3,3, 2,2,2, 1,1,1, 1,1,1);
  ConvGeom g_c2  = mkG(11,128,128,32, 11,128,128,32, 3,3,3, 1,1,1, 1,1,1, 1,1,1);
  ConvGeom g_md3 = mkG(11,128,128,1,  6,64,64,1,     3,3,3, 2,2,2, 1,1,1, 1,1,1);
  ConvGeom g_d3  = mkG(11,128,128,32, 6,64,64,64,    3,3,3, 2,2,2, 1,1,1, 1,1,1);
  ConvGeom g_c3  = mkG(6,64,64,64,    6,64,64,64,    3,3,3, 1,1,1, 1,1,1, 1,1,1);
  ConvGeom g_md4 = mkG(6,64,64,1,     2,32,32,1,     3,3,3, 2,2,2, 0,1,1, 1,1,1);
  ConvGeom g_d4  = mkG(6,64,64,64,    2,32,32,64,    3,3,3, 2,2,2, 0,1,1, 1,1,1);
  ConvGeom g_md4a= mkG(6,64,64,1,     2,32,32,1,     3,3,3, 2,2,2, 0,2,2, 1,2,2);
  ConvGeom g_d4a = mkG(6,64,64,64,    2,32,32,64,    3,3,3, 2,2,2, 0,2,2, 1,2,2);
  ConvGeom g_md4b= mkG(6,64,64,1,     2,32,32,1,     3,5,5, 2,2,2, 0,6,6, 1,3,3);
  ConvGeom g_d4b = mkG(6,64,64,64,    2,32,32,64,    3,5,5, 2,2,2, 0,6,6, 1,3,3);
  ConvGeom g_c4  = mkG(2,32,32,64,    2,32,32,64,    3,3,3, 1,1,1, 1,1,1, 1,1,1);
  ConvGeom g_md2d= mkG(1,32,32,1,     1,32,32,1,     1,3,3, 1,1,1, 0,1,1, 1,1,1);
  ConvGeom g_sh1 = mkG(1,32,32,64,    1,32,32,128,   1,3,3, 1,1,1, 0,1,1, 1,1,1);
  ConvGeom g_sh2 = mkG(1,32,32,128,   1,32,32,128,   1,3,3, 1,1,1, 0,1,1, 1,1,1);

  (void)hipMemsetAsync(grid0, 0, NV0 * 4 * sizeof(float), stream);
  (void)hipMemsetAsync(mask0, 0, NV0 * sizeof(float), stream);
  k_scatter<<<(N + 255) / 256, 256, 0, stream>>>(feat, coords, grid0, mask0, N, H0, W0);
  k_binarize<<<(int)((NV0 + 255) / 256), 256, 0, stream>>>(mask0, (int)NV0);

  runConv(grid0, wf_ci, bufA, g_ci);
  runBN(bufA, nullptr, mask0, P(CI_G), P(CI_B), bufB, NV0, 16);

  runBlock(bufB, bufA, bufC, wf_c1b0w1, wf_c1b0w2, mask0, C1 + 0, NV0, 16, g_c1);
  runBlock(bufC, bufA, bufB, wf_c1b1w1, wf_c1b1w2, mask0, C1 + 8, NV0, 16, g_c1);

  runDilate(mask0, m2, g_md2);
  runConv(bufB, wf_c2d, bufA, g_d2);
  runBN(bufA, nullptr, m2, P(C2 + 17), P(C2 + 16), bufC, NV2, 32);
  runBlock(bufC, bufA, bufB, wf_c2b0w1, wf_c2b0w2, m2, C2 + 0, NV2, 32, g_c2);
  runBlock(bufB, bufA, bufC, wf_c2b1w1, wf_c2b1w2, m2, C2 + 8, NV2, 32, g_c2);

  runDilate(m2, m3, g_md3);
  runConv(bufC, wf_c3d, bufA, g_d3);
  runBN(bufA, nullptr, m3, P(C3 + 17), P(C3 + 16), bufB, NV3, 64);
  runBlock(bufB, bufA, bufC, wf_c3b0w1, wf_c3b0w2, m3, C3 + 0, NV3, 64, g_c3);
  runBlock(bufC, bufA, bufB, wf_c3b1w1, wf_c3b1w2, m3, C3 + 8, NV3, 64, g_c3);
  // x3 in bufB

  auto runBranch = [&](int S, const unsigned short* wfd,
                       const unsigned short* b0w1, const unsigned short* b0w2,
                       const unsigned short* b1w1, const unsigned short* b1w2,
                       ConvGeom gd, ConvGeom gmd, float* mout, float* out) {
    runDilate(m3, mout, gmd);
    runConv(bufB, wfd, s1, gd);
    runBN(s1, nullptr, mout, P(S + 17), P(S + 16), s2, NV4, 64);
    runBlock(s2, s1, s3, b0w1, b0w2, mout, S + 0, NV4, 64, g_c4);
    runBlock(s3, s1, out, b1w1, b1w2, mout, S + 8, NV4, 64, g_c4);
  };
  runBranch(C4, wf_c4d, wf_c4b0w1, wf_c4b0w2, wf_c4b1w1, wf_c4b1w2, g_d4, g_md4, m4, b4o0);
  runBranch(C4A, wf_c4ad, wf_c4ab0w1, wf_c4ab0w2, wf_c4ab1w1, wf_c4ab1w2, g_d4a, g_md4a, m4a, b4o1);
  runBranch(C4B, wf_c4bd, wf_c4bb0w1, wf_c4bb0w2, wf_c4bb1w1, wf_c4bb1w2, g_d4b, g_md4b, m4b, b4o2);

  int n4 = (int)(NV4 * 64);
  k_add3<<<(n4 + 255) / 256, 256, 0, stream>>>(b4o0, b4o1, b4o2, xs, n4);
  k_max3<<<(int)((NV4 + 255) / 256), 256, 0, stream>>>(m4, m4a, m4b, msu, (int)NV4);
  k_hcompress<<<(HW4 * 64 + 255) / 256, 256, 0, stream>>>(
      xs, msu, x2d, m2d, 2, HW4, 64);

  runDilate(m2d, m2d1, g_md2d);
  runConv(x2d, wf_sh1, t2d, g_sh1);
  runBN(t2d, nullptr, m2d1, P(SH_G1), P(SH_B1), y2d, HW4, 128);
  runDilate(m2d1, m2d2, g_md2d);
  runConv(y2d, wf_sh2, t2d, g_sh2);
  runBN(t2d, nullptr, m2d2, P(SH_G2), P(SH_B2), (float*)d_out, HW4, 128);
}
